// Attention3D_75539884802652
// MI455X (gfx1250) — compile-verified, matrix operand order checked by reading, not run on hardware
//
#include <hip/hip_runtime.h>
#include <hip/hip_bf16.h>

#define HEADS 4
#define GROUPS 8

static constexpr int B_  = 2;
static constexpr int C_  = 256;
static constexpr int N_  = 4096;     // 16*16*16
static constexpr int HD  = 64;       // head dim
static constexpr float EPS = 1e-5f;

typedef __bf16 bf16_t;
typedef __attribute__((ext_vector_type(16))) __bf16 v16bf;
typedef __attribute__((ext_vector_type(8)))  __bf16 v8bf;
typedef __attribute__((ext_vector_type(8)))  float  v8f;
typedef __attribute__((ext_vector_type(4)))  float  v4f;
typedef __attribute__((ext_vector_type(4)))  int    v4i;

#define WMMA_BF16(a, b, c) \
    __builtin_amdgcn_wmma_f32_16x16x32_bf16(false, (a), false, (b), (short)0, (c), false, false)

// --------------------------------------------------------------------------
// Async global->LDS copy (CDNA5). Guarded so the build never breaks.
// --------------------------------------------------------------------------
#if defined(__has_builtin)
#  if __has_builtin(__builtin_amdgcn_global_load_async_to_lds_b128)
#    define HAVE_ASYNC 1
#  endif
#endif
#ifndef HAVE_ASYNC
#  define HAVE_ASYNC 0
#endif

typedef __attribute__((address_space(1))) v4i* glb_v4i_p;
typedef __attribute__((address_space(3))) v4i* lds_v4i_p;

__device__ __forceinline__ void async_cp16(const bf16_t* g, bf16_t* l) {
#if HAVE_ASYNC
    __builtin_amdgcn_global_load_async_to_lds_b128(
        (glb_v4i_p)(void*)g, (lds_v4i_p)(void*)l, 0, 0);
#else
    *(v8bf*)l = *(const v8bf*)g;
#endif
}

__device__ __forceinline__ void wait_async0() {
#if HAVE_ASYNC
#  if __has_builtin(__builtin_amdgcn_s_wait_asynccnt)
    __builtin_amdgcn_s_wait_asynccnt(0);
#  else
    asm volatile("s_wait_asynccnt 0" ::: "memory");
#  endif
#endif
}

// --------------------------------------------------------------------------
// 16x32 bf16 A/B fragment from a K-contiguous row:
// element h -> K = hiOff + (h<8 ? h : 16+(h-8))  => two 16-byte loads.
// --------------------------------------------------------------------------
__device__ __forceinline__ v16bf load_frag(const bf16_t* row, int hiOff) {
    v8bf lo = *(const v8bf*)(row + hiOff);
    v8bf hi = *(const v8bf*)(row + hiOff + 16);
    return __builtin_shufflevector(lo, hi, 0, 1, 2, 3, 4, 5, 6, 7,
                                           8, 9, 10, 11, 12, 13, 14, 15);
}

// --------------------------------------------------------------------------
// Kernel 0: fp32 -> bf16 weight conversion.
// --------------------------------------------------------------------------
__global__ __launch_bounds__(256) void cvt_bf16(const float* __restrict__ src,
                                                bf16_t* __restrict__ dst, int n) {
    int i = blockIdx.x * 256 + threadIdx.x;
    if (i < n) dst[i] = (bf16_t)src[i];
}

// --------------------------------------------------------------------------
// Kernel 1: GroupNorm statistics. grid = B*GROUPS, 256 threads.
// --------------------------------------------------------------------------
__global__ __launch_bounds__(256) void gn_stats(const float* __restrict__ x,
                                                float* __restrict__ mean_out,
                                                float* __restrict__ rstd_out) {
    int bg = blockIdx.x;
    const int total4 = (C_ / GROUPS) * N_ / 4;            // 32768 float4
    const v4f* xp = (const v4f*)(x + (long)bg * (C_ / GROUPS) * N_);
    float s = 0.f, ss = 0.f;
    for (int i = threadIdx.x; i < total4; i += 256) {
        v4f v = xp[i];
        s  += v.x + v.y + v.z + v.w;
        ss += v.x * v.x + v.y * v.y + v.z * v.z + v.w * v.w;
    }
    __shared__ float sh_s[256], sh_q[256];
    sh_s[threadIdx.x] = s; sh_q[threadIdx.x] = ss;
    __syncthreads();
    for (int off = 128; off > 0; off >>= 1) {
        if (threadIdx.x < off) {
            sh_s[threadIdx.x] += sh_s[threadIdx.x + off];
            sh_q[threadIdx.x] += sh_q[threadIdx.x + off];
        }
        __syncthreads();
    }
    if (threadIdx.x == 0) {
        float tot = (float)(C_ / GROUPS) * N_;
        float m   = sh_s[0] / tot;
        float var = sh_q[0] / tot - m * m;
        mean_out[bg] = m;
        rstd_out[bg] = rsqrtf(var + EPS);
    }
}

// --------------------------------------------------------------------------
// Kernel 2: GroupNorm apply with LDS transpose.
// Coalesced v4f reads of x[b][c][n]; coalesced b16 writes of xn[b][n][c].
// grid = B * (N/64); block tile = 256 c x 64 n.
// --------------------------------------------------------------------------
__global__ __launch_bounds__(256) void gn_apply(const float* __restrict__ x,
                                                const float* __restrict__ gn_w,
                                                const float* __restrict__ gn_b,
                                                const float* __restrict__ mean,
                                                const float* __restrict__ rstd,
                                                bf16_t* __restrict__ xn) {
    __shared__ float scs[C_], shs[C_];
    __shared__ bf16_t tile[C_][68];            // 64 n + 4 pad
    int tid = threadIdx.x;
    int b   = blockIdx.x >> 6;                 // 64 n-tiles per batch
    int n0  = (blockIdx.x & 63) * 64;
    {
        int c  = tid;
        int bg = b * GROUPS + (c >> 5);
        float w = gn_w[c] * rstd[bg];
        scs[c] = w;
        shs[c] = gn_b[c] - mean[bg] * w;
    }
    __syncthreads();
    #pragma unroll
    for (int k = 0; k < 16; ++k) {             // 4096 v4f chunks
        int id = k * 256 + tid;
        int c = id >> 4, n4 = (id & 15) * 4;
        v4f v = *(const v4f*)(x + ((long)b * C_ + c) * N_ + n0 + n4);
        float sc = scs[c], sh = shs[c];
        bf16_t* t = &tile[c][n4];
        t[0] = (bf16_t)(v.x * sc + sh);
        t[1] = (bf16_t)(v.y * sc + sh);
        t[2] = (bf16_t)(v.z * sc + sh);
        t[3] = (bf16_t)(v.w * sc + sh);
    }
    __syncthreads();
    #pragma unroll 8
    for (int nn = 0; nn < 64; ++nn)            // coalesced over c = tid
        xn[((long)b * N_ + n0 + nn) * C_ + tid] = tile[tid][nn];
}

// --------------------------------------------------------------------------
// Kernel 3: QKV GEMM, 2x2 tiles per wave (32 o x 32 n), K=256 in 8 steps.
// 4 WMMAs per 4 fragment loads. q,k -> [b][h][n][64]; v -> [b][h][d][n].
// --------------------------------------------------------------------------
__global__ __launch_bounds__(256) void qkv_gemm(const bf16_t* __restrict__ W,
                                                const float* __restrict__ bias,
                                                const bf16_t* __restrict__ xn,
                                                bf16_t* __restrict__ qd,
                                                bf16_t* __restrict__ kd,
                                                bf16_t* __restrict__ vd) {
    int lane = threadIdx.x & 31;
    int wave = threadIdx.x >> 5;
    int tile = blockIdx.x * 8 + wave;
    const int Mt = 768 / 32, Nt = N_ / 32;     // 24, 128
    int bidx  = tile / (Mt * Nt);
    int t2    = tile % (Mt * Nt);
    int oBase = (t2 / Nt) * 32;
    int nBase = (t2 % Nt) * 32;
    int hiOff = (lane & 16) ? 8 : 0;

    const bf16_t* W0 = W + (long)(oBase + (lane & 15)) * C_;
    const bf16_t* W1 = W0 + 16 * C_;
    const bf16_t* X0 = xn + ((long)bidx * N_ + nBase + (lane & 15)) * C_;
    const bf16_t* X1 = X0 + 16 * C_;

    v8f zero8 = {};
    v8f acc[2][2];
    acc[0][0] = zero8; acc[0][1] = zero8; acc[1][0] = zero8; acc[1][1] = zero8;
    #pragma unroll
    for (int k0 = 0; k0 < C_; k0 += 32) {
        v16bf a0 = load_frag(W0 + k0, hiOff);
        v16bf a1 = load_frag(W1 + k0, hiOff);
        v16bf b0 = load_frag(X0 + k0, hiOff);
        v16bf b1 = load_frag(X1 + k0, hiOff);
        acc[0][0] = WMMA_BF16(a0, b0, acc[0][0]);
        acc[0][1] = WMMA_BF16(a0, b1, acc[0][1]);
        acc[1][0] = WMMA_BF16(a1, b0, acc[1][0]);
        acc[1][1] = WMMA_BF16(a1, b1, acc[1][1]);
    }

    int col = lane & 15, rowAdd = (lane >> 4) * 8;
    #pragma unroll
    for (int mi = 0; mi < 2; ++mi)
        #pragma unroll
        for (int ni = 0; ni < 2; ++ni) {
            int o0 = oBase + mi * 16 + rowAdd;   // 8 consecutive outputs
            int n  = nBase + ni * 16 + col;
            v8bf pk;
            #pragma unroll
            for (int r = 0; r < 8; ++r) pk[r] = (bf16_t)(acc[mi][ni][r] + bias[o0 + r]);
            int which = o0 >> 8;                 // 0=q 1=k 2=v
            int c0 = o0 & 255;
            int h = c0 >> 6, d0 = c0 & 63;
            if (which == 0) {
                *(v8bf*)(qd + (((long)bidx * HEADS + h) * N_ + n) * HD + d0) = pk;
            } else if (which == 1) {
                *(v8bf*)(kd + (((long)bidx * HEADS + h) * N_ + n) * HD + d0) = pk;
            } else {                             // v: [b][h][d][n]
                #pragma unroll
                for (int r = 0; r < 8; ++r)
                    vd[(((long)bidx * HEADS + h) * HD + d0 + r) * N_ + n] = pk[r];
            }
        }
}

// --------------------------------------------------------------------------
// Kernel 4: flash attention.  grid = B*H*(N/128), 8 waves, 16 q-rows/wave.
// K/V tiles async-copied to LDS (double-buffered); online softmax;
// P restaged via per-wave LDS for the O GEMM.
// --------------------------------------------------------------------------
__global__ __launch_bounds__(256) void attn_kernel(const bf16_t* __restrict__ qd,
                                                   const bf16_t* __restrict__ kd,
                                                   const bf16_t* __restrict__ vd,
                                                   bf16_t* __restrict__ omat) {
    __shared__ bf16_t kT[2][32 * 64];     // [buf][key m][d]
    __shared__ bf16_t vT[2][64 * 32];     // [buf][d][key m]
    __shared__ bf16_t pB[8 * 16 * 32];    // per-wave P tile [row][m]
    __shared__ bf16_t oB[8 * 16 * 64];    // per-wave O tile [row][d]

    int tid  = threadIdx.x;
    int lane = tid & 31;
    int wave = tid >> 5;
    int qb   = blockIdx.x & 31;
    int bh   = blockIdx.x >> 5;           // b*HEADS + head
    int b    = bh / HEADS, head = bh % HEADS;
    const long base = (long)bh * N_ * HD;
    int qRow = qb * 128 + wave * 16;
    const float scale = 0.125f;           // 64^-0.5
    int hiOff = (lane & 16) ? 8 : 0;

    const bf16_t* kbase = kd + base;      // [n][d]
    const bf16_t* vbase = vd + base;      // [d][n]

    v16bf qf[2];
    {
        const bf16_t* qp = qd + base + (long)(qRow + (lane & 15)) * HD;
        qf[0] = load_frag(qp, hiOff);
        qf[1] = load_frag(qp + 32, hiOff);
    }

    v8f zero8 = {};
    v8f oacc[4];
    #pragma unroll
    for (int t = 0; t < 4; ++t) oacc[t] = zero8;
    float m_r[8], l_r[8];
    #pragma unroll
    for (int r = 0; r < 8; ++r) { m_r[r] = -3.0e38f; l_r[r] = 0.f; }

    // per-thread staging chunks: one b128 for K, one b128 for V, per tile
    int km = tid >> 3, koff = (tid & 7) * 8;      // K: 32 rows x 8 chunks
    int vdr = tid >> 2, voff = (tid & 3) * 8;     // V: 64 rows x 4 chunks

    const int NT = N_ / 32;
    async_cp16(kbase + (long)km * HD + koff,  &kT[0][km * 64 + koff]);
    async_cp16(vbase + (long)vdr * N_ + voff, &vT[0][vdr * 32 + voff]);

    for (int j = 0; j < NT; ++j) {
        wait_async0();
        __syncthreads();                  // tile j visible; buf (j+1)&1 free
        if (j + 1 < NT) {
            int jn = (j + 1) * 32;
            int nb = (j + 1) & 1;
            async_cp16(kbase + (long)(jn + km) * HD + koff, &kT[nb][km * 64 + koff]);
            async_cp16(vbase + (long)vdr * N_ + jn + voff,  &vT[nb][vdr * 32 + voff]);
        }
        const bf16_t* kTile = kT[j & 1];
        const bf16_t* vTile = vT[j & 1];

        // S tiles: 16 q-rows x 32 keys (two 16-col tiles)
        v8f s0 = zero8, s1 = zero8;
        #pragma unroll
        for (int t = 0; t < 2; ++t) {
            v16bf b0 = load_frag(kTile + (lane & 15) * 64 + t * 32, hiOff);
            v16bf b1 = load_frag(kTile + (16 + (lane & 15)) * 64 + t * 32, hiOff);
            s0 = WMMA_BF16(qf[t], b0, s0);
            s1 = WMMA_BF16(qf[t], b1, s1);
        }

        // online softmax: VGPR r -> lanes 0-15 row r, lanes 16-31 row r+8
        float corr[8];
        bf16_t* pw = &pB[wave * 512];
        int lrow8 = (lane >> 4) * 8;
        #pragma unroll
        for (int r = 0; r < 8; ++r) {
            float a0 = s0[r] * scale, a1 = s1[r] * scale;
            float mx = fmaxf(a0, a1);
            #pragma unroll
            for (int off = 1; off < 16; off <<= 1)
                mx = fmaxf(mx, __shfl_xor(mx, off, 32));
            float mn = fmaxf(m_r[r], mx);
            float cf = __expf(m_r[r] - mn);
            float p0 = __expf(a0 - mn), p1 = __expf(a1 - mn);
            float ps = p0 + p1;
            #pragma unroll
            for (int off = 1; off < 16; off <<= 1)
                ps += __shfl_xor(ps, off, 32);
            l_r[r] = l_r[r] * cf + ps;
            m_r[r] = mn;
            corr[r] = cf;
            pw[(r + lrow8) * 32 + (lane & 15)]      = (bf16_t)p0;
            pw[(r + lrow8) * 32 + 16 + (lane & 15)] = (bf16_t)p1;
        }
        #pragma unroll
        for (int t = 0; t < 4; ++t)
            #pragma unroll
            for (int r = 0; r < 8; ++r) oacc[t][r] *= corr[r];

        // O += P(16x32) * V(32x64); same-wave LDS W->R is DS-ordered
        v16bf pf = load_frag(&pB[wave * 512 + (lane & 15) * 32], hiOff);
        #pragma unroll
        for (int t = 0; t < 4; ++t) {
            v16bf vf = load_frag(vTile + (t * 16 + (lane & 15)) * 32, hiOff);
            oacc[t] = WMMA_BF16(pf, vf, oacc[t]);
        }
        __syncthreads();                  // done reading buf j&1
    }

    // normalize, stage O tile in LDS, store coalesced b128 to [b][n][c]
    float inv[8];
    #pragma unroll
    for (int r = 0; r < 8; ++r) inv[r] = 1.f / l_r[r];
    int col = lane & 15, rowAdd = (lane >> 4) * 8;
    bf16_t* ow = &oB[wave * 1024];
    #pragma unroll
    for (int t = 0; t < 4; ++t)
        #pragma unroll
        for (int r = 0; r < 8; ++r)
            ow[(r + rowAdd) * 64 + t * 16 + col] = (bf16_t)(oacc[t][r] * inv[r]);
    #pragma unroll
    for (int ck = 0; ck < 4; ++ck) {
        int idx  = ck * 256 + lane * 8;   // within 16x64 wave tile
        int lrow = idx >> 6, dd = idx & 63;
        int n = qRow + lrow;
        *(v8bf*)(omat + ((long)b * N_ + n) * C_ + head * HD + dd) =
            *(const v8bf*)(ow + idx);
    }
}

// --------------------------------------------------------------------------
// Kernel 5: projection GEMM + bias + residual.  Roles swapped:
// M = n (rows), N = o (cols) so each lane owns 8 consecutive n -> vector
// residual loads + stores on fp32 [b][c][n].  2x2 tiles per wave.
// --------------------------------------------------------------------------
__global__ __launch_bounds__(256) void proj_gemm(const bf16_t* __restrict__ W,
                                                 const float* __restrict__ bias,
                                                 const bf16_t* __restrict__ ain,
                                                 const float* __restrict__ x,
                                                 float* __restrict__ out) {
    int lane = threadIdx.x & 31;
    int wave = threadIdx.x >> 5;
    int tile = blockIdx.x * 8 + wave;
    const int Mt = N_ / 32, Nt = C_ / 32;      // 128, 8
    int bidx  = tile / (Mt * Nt);
    int t2    = tile % (Mt * Nt);
    int nBase = (t2 / Nt) * 32;
    int oBase = (t2 % Nt) * 32;
    int hiOff = (lane & 16) ? 8 : 0;

    const bf16_t* A0 = ain + ((long)bidx * N_ + nBase + (lane & 15)) * C_;
    const bf16_t* A1 = A0 + 16 * C_;
    const bf16_t* B0 = W + (long)(oBase + (lane & 15)) * C_;
    const bf16_t* B1 = B0 + 16 * C_;

    v8f zero8 = {};
    v8f acc[2][2];
    acc[0][0] = zero8; acc[0][1] = zero8; acc[1][0] = zero8; acc[1][1] = zero8;
    #pragma unroll
    for (int k0 = 0; k0 < C_; k0 += 32) {
        v16bf a0 = load_frag(A0 + k0, hiOff);
        v16bf a1 = load_frag(A1 + k0, hiOff);
        v16bf b0 = load_frag(B0 + k0, hiOff);
        v16bf b1 = load_frag(B1 + k0, hiOff);
        acc[0][0] = WMMA_BF16(a0, b0, acc[0][0]);
        acc[0][1] = WMMA_BF16(a0, b1, acc[0][1]);
        acc[1][0] = WMMA_BF16(a1, b0, acc[1][0]);
        acc[1][1] = WMMA_BF16(a1, b1, acc[1][1]);
    }

    int col = lane & 15, rowAdd = (lane >> 4) * 8;
    #pragma unroll
    for (int mi = 0; mi < 2; ++mi)
        #pragma unroll
        for (int ni = 0; ni < 2; ++ni) {
            int o  = oBase + ni * 16 + col;
            int n0 = nBase + mi * 16 + rowAdd;   // 8 consecutive n
            float bo = bias[o];
            long idx = ((long)bidx * C_ + o) * (long)N_ + n0;
            v4f x0 = *(const v4f*)(x + idx);
            v4f x1 = *(const v4f*)(x + idx + 4);
            v4f r0, r1;
            r0.x = acc[mi][ni][0] + bo + x0.x;
            r0.y = acc[mi][ni][1] + bo + x0.y;
            r0.z = acc[mi][ni][2] + bo + x0.z;
            r0.w = acc[mi][ni][3] + bo + x0.w;
            r1.x = acc[mi][ni][4] + bo + x1.x;
            r1.y = acc[mi][ni][5] + bo + x1.y;
            r1.z = acc[mi][ni][6] + bo + x1.z;
            r1.w = acc[mi][ni][7] + bo + x1.w;
            *(v4f*)(out + idx)     = r0;
            *(v4f*)(out + idx + 4) = r1;
        }
}

// --------------------------------------------------------------------------
extern "C" void kernel_launch(void* const* d_in, const int* in_sizes, int n_in,
                              void* d_out, int out_size, void* d_ws, size_t ws_size,
                              hipStream_t stream) {
    (void)in_sizes; (void)n_in; (void)out_size; (void)ws_size;
    const float* x      = (const float*)d_in[0];
    const float* gn_w   = (const float*)d_in[1];
    const float* gn_b   = (const float*)d_in[2];
    const float* qkv_w  = (const float*)d_in[3];
    const float* qkv_b  = (const float*)d_in[4];
    const float* proj_w = (const float*)d_in[5];
    const float* proj_b = (const float*)d_in[6];
    float* out = (float*)d_out;

    char* ws = (char*)d_ws;
    float* mean = (float*)ws;                    // 16 floats
    float* rstd = (float*)(ws + 64);             // 16 floats
    bf16_t* xn  = (bf16_t*)(ws + 256);           // [b][n][c]
    const size_t BCN = (size_t)B_ * C_ * N_;     // 2,097,152 elems
    bf16_t* qb  = xn + BCN;                      // [b][h][n][64]
    bf16_t* kb  = qb + BCN;                      // [b][h][n][64]
    bf16_t* vb  = kb + BCN;                      // [b][h][d][n]
    bf16_t* ao  = vb + BCN;                      // [b][n][c]
    bf16_t* wqb = ao + BCN;                      // bf16 qkv_w [768][256]
    bf16_t* wpb = wqb + 768 * C_;                // bf16 proj_w [256][256]

    cvt_bf16 <<<(768 * C_) / 256,                 256, 0, stream>>>(qkv_w, wqb, 768 * C_);
    cvt_bf16 <<<(C_ * C_) / 256,                  256, 0, stream>>>(proj_w, wpb, C_ * C_);
    gn_stats <<<B_ * GROUPS,                      256, 0, stream>>>(x, mean, rstd);
    gn_apply <<<B_ * (N_ / 64),                   256, 0, stream>>>(x, gn_w, gn_b, mean, rstd, xn);
    qkv_gemm <<<(B_ * (768/32) * (N_/32)) / 8,    256, 0, stream>>>(wqb, qkv_b, xn, qb, kb, vb);
    attn_kernel<<<B_ * HEADS * (N_ / 128),        256, 0, stream>>>(qb, kb, vb, ao);
    proj_gemm<<<(B_ * (N_/32) * (C_/32)) / 8,     256, 0, stream>>>(wpb, proj_b, ao, x, out);
}
